// Block_39728447488136
// MI455X (gfx1250) — matrix operand order repaired and checked by reading, not compiled, not run on hardware
//
#include <hip/hip_runtime.h>

// ---------------------------------------------------------------------------
// Transformer block for MI455X (gfx1250, wave32, WMMA 16x16x32 f16->f32)
// B=4, T=2048, D=1024, H=16, HS=64, FF=4096
// ---------------------------------------------------------------------------

#define Bv  4
#define Tv  2048
#define Dv  1024
#define Hv  16
#define HSv 64
#define Fv  4096
#define ROWS (Bv * Tv)          // 8192

typedef __attribute__((ext_vector_type(16))) _Float16 h16v;
typedef __attribute__((ext_vector_type(8)))  _Float16 h8v;
typedef __attribute__((ext_vector_type(4)))  _Float16 h4v;
typedef __attribute__((ext_vector_type(8)))  float    f8v;

union AFrag { h16v v; h8v h[2]; };

__device__ __forceinline__ f8v wmma_f16(h16v a, h16v b, f8v c) {
    // v_wmma_f32_16x16x32_f16 (8-arg form: neg_a, A, neg_b, B, c_mod, C, reuse_a, reuse_b)
    return __builtin_amdgcn_wmma_f32_16x16x32_f16(false, a, false, b, (short)0, c,
                                                  false, false);
}

// A-matrix fragment (16x32 f16), row = lane&15, K contiguous at k0+16*(lane>>4) .. +15
__device__ __forceinline__ h16v load_a(const _Float16* __restrict__ row_ptr, int lq) {
    AFrag a;
    a.h[0] = *(const h8v*)(row_ptr + 16 * lq);
    a.h[1] = *(const h8v*)(row_ptr + 16 * lq + 8);
    return a.v;
}

// ---------------------------------------------------------------------------
// LayerNorm: per-row mean/var, writes f32 and f16 copies. 256 thr, D=1024.
// ---------------------------------------------------------------------------
__global__ __launch_bounds__(256) void ln_kernel(const float* __restrict__ x,
                                                 const float* __restrict__ g,
                                                 const float* __restrict__ be,
                                                 float* __restrict__ hf,
                                                 _Float16* __restrict__ hh) {
    __shared__ float s_sum[8], s_sq[8];
    const int row = blockIdx.x;
    const int tid = threadIdx.x;
    const float* xr = x + (size_t)row * Dv;
    float4 v = *(const float4*)(xr + tid * 4);
    float s = v.x + v.y + v.z + v.w;
    float q = v.x * v.x + v.y * v.y + v.z * v.z + v.w * v.w;
#pragma unroll
    for (int off = 16; off >= 1; off >>= 1) {
        s += __shfl_xor(s, off, 32);
        q += __shfl_xor(q, off, 32);
    }
    if ((tid & 31) == 0) { s_sum[tid >> 5] = s; s_sq[tid >> 5] = q; }
    __syncthreads();
    float ts = 0.f, tq = 0.f;
#pragma unroll
    for (int i = 0; i < 8; ++i) { ts += s_sum[i]; tq += s_sq[i]; }
    const float mean = ts * (1.0f / Dv);
    const float var  = tq * (1.0f / Dv) - mean * mean;
    const float rstd = rsqrtf(var + 1e-5f);

    const int n0 = tid * 4;
    float o0 = (v.x - mean) * rstd * g[n0 + 0] + be[n0 + 0];
    float o1 = (v.y - mean) * rstd * g[n0 + 1] + be[n0 + 1];
    float o2 = (v.z - mean) * rstd * g[n0 + 2] + be[n0 + 2];
    float o3 = (v.w - mean) * rstd * g[n0 + 3] + be[n0 + 3];
    float4 of = {o0, o1, o2, o3};
    *(float4*)(hf + (size_t)row * Dv + n0) = of;
    h4v oh = {(_Float16)o0, (_Float16)o1, (_Float16)o2, (_Float16)o3};
    *(h4v*)(hh + (size_t)row * Dv + n0) = oh;
}

// ---------------------------------------------------------------------------
// Convert f32 W[K][N] (row-major) -> f16 WT[N][K] (N-major) so WMMA B-fragments
// are contiguous 32B loads. batch via blockIdx.y.
// ---------------------------------------------------------------------------
__global__ __launch_bounds__(256) void cvt_transpose(const float* __restrict__ W,
                                                     _Float16* __restrict__ WT,
                                                     int K, int N) {
    const size_t batch_off = (size_t)blockIdx.y * K * N;
    int idx = blockIdx.x * 256 + threadIdx.x;
    if (idx < K * N) {
        int k = idx / N, n = idx % N;
        WT[batch_off + (size_t)n * K + k] = (_Float16)W[batch_off + idx];
    }
}

// ---------------------------------------------------------------------------
// QKV: per (b,h), C[T,64] = h16[b] @ W{q,k,v}T[h].  8 waves/block, each wave a
// 16x64 strip of all three outputs (12 accumulators, A reused 12x per K-step).
// V is stored transposed: vT[(bh*64+n)*T + t].
// ---------------------------------------------------------------------------
__global__ __launch_bounds__(256) void qkv_kernel(const _Float16* __restrict__ h16,
                                                  const _Float16* __restrict__ WqT,
                                                  const _Float16* __restrict__ WkT,
                                                  const _Float16* __restrict__ WvT,
                                                  _Float16* __restrict__ q16,
                                                  _Float16* __restrict__ k16,
                                                  _Float16* __restrict__ vT16) {
    const int bh = blockIdx.y;
    const int b = bh / Hv, h = bh % Hv;
    const int wave = threadIdx.x >> 5, lane = threadIdx.x & 31;
    const int lh = lane & 15, lq = lane >> 4;
    const int t0 = blockIdx.x * 128 + wave * 16;

    const _Float16* arow = h16 + (size_t)(b * Tv + t0 + lh) * Dv;
    const _Float16* wq0 = WqT + (size_t)h * HSv * Dv;
    const _Float16* wk0 = WkT + (size_t)h * HSv * Dv;
    const _Float16* wv0 = WvT + (size_t)h * HSv * Dv;

    f8v aq[4] = {}, ak[4] = {}, av[4] = {};
    for (int k0 = 0; k0 < Dv; k0 += 32) {
        h16v a = load_a(arow + k0, lq);
        __builtin_prefetch(arow + k0 + 256, 0, 1);   // next A cachelines
        const int boff = k0 + 16 * lq;
#pragma unroll
        for (int nt = 0; nt < 4; ++nt) {
            const size_t brow = (size_t)(nt * 16 + lh) * Dv + boff;
            aq[nt] = wmma_f16(a, *(const h16v*)(wq0 + brow), aq[nt]);
            ak[nt] = wmma_f16(a, *(const h16v*)(wk0 + brow), ak[nt]);
            av[nt] = wmma_f16(a, *(const h16v*)(wv0 + brow), av[nt]);
        }
    }
#pragma unroll
    for (int nt = 0; nt < 4; ++nt) {
        const int n = nt * 16 + lh;
#pragma unroll
        for (int r = 0; r < 8; ++r) {
            const int t = t0 + r + 8 * lq;
            q16[(size_t)(bh * Tv + t) * HSv + n] = (_Float16)aq[nt][r];
            k16[(size_t)(bh * Tv + t) * HSv + n] = (_Float16)ak[nt][r];
        }
        h8v vv;
#pragma unroll
        for (int r = 0; r < 8; ++r) vv[r] = (_Float16)av[nt][r];
        *(h8v*)(vT16 + (size_t)(bh * HSv + n) * Tv + t0 + 8 * lq) = vv;
    }
}

// ---------------------------------------------------------------------------
// Causal flash attention. One wave per 16-row query tile; stream 32-key blocks.
// scores = q@k^T * D^-0.5 -> online softmax -> P (LDS, f16) -> @ vT.
// ---------------------------------------------------------------------------
__global__ __launch_bounds__(256) void attn_kernel(const _Float16* __restrict__ q16,
                                                   const _Float16* __restrict__ k16,
                                                   const _Float16* __restrict__ vT16,
                                                   _Float16* __restrict__ o16) {
    __shared__ _Float16 Plds[8][16][32];       // per-wave 16x32 P tile, 8 KB
    const int bh = blockIdx.y;
    const int b = bh / Hv, h = bh % Hv;
    const int wave = threadIdx.x >> 5, lane = threadIdx.x & 31;
    const int lh = lane & 15, lq = lane >> 4;
    const int qBase = (blockIdx.x * 8 + wave) * 16;

    // Q fragments (K=64 -> two 16x32 A-fragments), held in registers.
    const _Float16* qrow = q16 + (size_t)(bh * Tv + qBase + lh) * HSv;
    h16v aqf[2];
    aqf[0] = load_a(qrow, lq);
    aqf[1] = load_a(qrow + 32, lq);

    f8v acc[4] = {};
    float mrow[8], lrow[8];
#pragma unroll
    for (int r = 0; r < 8; ++r) { mrow[r] = -3.0e38f; lrow[r] = 0.f; }

    const float scale = 0.03125f;              // D^-0.5 = 1/sqrt(1024)
    const int jmax = qBase + 16;               // exclusive (causal)

    for (int j0 = 0; j0 < jmax; j0 += 32) {
        // two 16x16 score tiles (keys j0..j0+15, j0+16..j0+31)
        f8v c0 = {}, c1 = {};
#pragma unroll
        for (int kc = 0; kc < 2; ++kc) {
            const int feat = kc * 32 + 16 * lq;
            h16v bk0 = *(const h16v*)(k16 + (size_t)(bh * Tv + j0 + lh) * HSv + feat);
            h16v bk1 = *(const h16v*)(k16 + (size_t)(bh * Tv + j0 + 16 + lh) * HSv + feat);
            c0 = wmma_f16(aqf[kc], bk0, c0);
            c1 = wmma_f16(aqf[kc], bk1, c1);
        }

        float corrv[8];
#pragma unroll
        for (int r = 0; r < 8; ++r) {
            const int qi = qBase + r + 8 * lq;
            float s0 = c0[r] * scale;
            float s1 = c1[r] * scale;
            if (j0 + lh > qi)      s0 = -3.0e38f;
            if (j0 + 16 + lh > qi) s1 = -3.0e38f;
            float mx = fmaxf(s0, s1);
#pragma unroll
            for (int off = 1; off <= 8; off <<= 1) mx = fmaxf(mx, __shfl_xor(mx, off, 32));
            const float mn = fmaxf(mrow[r], mx);
            const float p0 = __expf(s0 - mn);
            const float p1 = __expf(s1 - mn);
            float rs = p0 + p1;
#pragma unroll
            for (int off = 1; off <= 8; off <<= 1) rs += __shfl_xor(rs, off, 32);
            const float corr = __expf(mrow[r] - mn);
            lrow[r] = lrow[r] * corr + rs;
            mrow[r] = mn;
            corrv[r] = corr;
            Plds[wave][r + 8 * lq][lh]      = (_Float16)p0;
            Plds[wave][r + 8 * lq][lh + 16] = (_Float16)p1;
        }
#pragma unroll
        for (int nt = 0; nt < 4; ++nt)
#pragma unroll
            for (int r = 0; r < 8; ++r) acc[nt][r] *= corrv[r];

        // wave-private LDS RAW: wait for DS stores before A-fragment reload
        asm volatile("s_wait_dscnt 0" ::: "memory");

        AFrag ap;
        ap.h[0] = *(const h8v*)(&Plds[wave][lh][16 * lq]);
        ap.h[1] = *(const h8v*)(&Plds[wave][lh][16 * lq + 8]);
#pragma unroll
        for (int nt = 0; nt < 4; ++nt) {
            h16v bv = *(const h16v*)(vT16 + (size_t)(bh * HSv + nt * 16 + lh) * Tv
                                     + j0 + 16 * lq);
            acc[nt] = wmma_f16(ap.v, bv, acc[nt]);
        }
    }

#pragma unroll
    for (int nt = 0; nt < 4; ++nt) {
        const int n = h * HSv + nt * 16 + lh;
#pragma unroll
        for (int r = 0; r < 8; ++r) {
            const int qi = qBase + r + 8 * lq;
            o16[(size_t)(b * Tv + qi) * Dv + n] = (_Float16)(acc[nt][r] / lrow[r]);
        }
    }
}

// ---------------------------------------------------------------------------
// Generic WMMA GEMM: C = A(MxK,f16) @ BT(NxK,f16)^T + bias [+resid][ReLU].
// 8 waves/block -> 64x256 block tile; wave tile 32x64 (two A-frags share each
// B-frag: 8 WMMAs per 6x32B loads per K-step).
// ---------------------------------------------------------------------------
template <bool RELU, bool RESID, bool OUT16>
__global__ __launch_bounds__(256) void gemm_kernel(const _Float16* __restrict__ A,
                                                   const _Float16* __restrict__ BT,
                                                   const float* __restrict__ bias,
                                                   const float* __restrict__ resid,
                                                   float* __restrict__ Cf,
                                                   _Float16* __restrict__ Ch,
                                                   int M, int N, int K) {
    const int wave = threadIdx.x >> 5, lane = threadIdx.x & 31;
    const int lh = lane & 15, lq = lane >> 4;
    const int m0 = blockIdx.y * 64 + (wave & 1) * 32;
    const int n0 = blockIdx.x * 256 + (wave >> 1) * 64;

    const _Float16* arow0 = A + (size_t)(m0 + lh) * K;
    const _Float16* arow1 = A + (size_t)(m0 + 16 + lh) * K;
    f8v acc[2][4] = {};
    for (int k0 = 0; k0 < K; k0 += 32) {
        h16v a0 = load_a(arow0 + k0, lq);
        h16v a1 = load_a(arow1 + k0, lq);
        __builtin_prefetch(arow0 + k0 + 256, 0, 1);
        __builtin_prefetch(arow1 + k0 + 256, 0, 1);
        const int boff = k0 + 16 * lq;
#pragma unroll
        for (int nt = 0; nt < 4; ++nt) {
            h16v bf = *(const h16v*)(BT + (size_t)(n0 + nt * 16 + lh) * K + boff);
            acc[0][nt] = wmma_f16(a0, bf, acc[0][nt]);
            acc[1][nt] = wmma_f16(a1, bf, acc[1][nt]);
        }
    }
#pragma unroll
    for (int mi = 0; mi < 2; ++mi) {
#pragma unroll
        for (int nt = 0; nt < 4; ++nt) {
            const int n = n0 + nt * 16 + lh;
            const float bn = bias[n];
#pragma unroll
            for (int r = 0; r < 8; ++r) {
                const int m = m0 + mi * 16 + r + 8 * lq;
                float v = acc[mi][nt][r] + bn;
                if constexpr (RESID) v += resid[(size_t)m * N + n];
                if constexpr (RELU) v = fmaxf(v, 0.0f);
                if constexpr (OUT16) Ch[(size_t)m * N + n] = (_Float16)v;
                else                 Cf[(size_t)m * N + n] = v;
            }
        }
    }
}

// ---------------------------------------------------------------------------
// Host-side orchestration
// ---------------------------------------------------------------------------
extern "C" void kernel_launch(void* const* d_in, const int* in_sizes, int n_in,
                              void* d_out, int out_size, void* d_ws, size_t ws_size,
                              hipStream_t stream) {
    const float* x   = (const float*)d_in[0];
    const float* Wq  = (const float*)d_in[1];
    const float* Wk  = (const float*)d_in[2];
    const float* Wv  = (const float*)d_in[3];
    const float* Wp  = (const float*)d_in[4];
    const float* bp  = (const float*)d_in[5];
    const float* W1  = (const float*)d_in[6];
    const float* b1  = (const float*)d_in[7];
    const float* W2  = (const float*)d_in[8];
    const float* b2  = (const float*)d_in[9];
    const float* g1  = (const float*)d_in[10];
    const float* be1 = (const float*)d_in[11];
    const float* g2  = (const float*)d_in[12];
    const float* be2 = (const float*)d_in[13];
    float* out = (float*)d_out;

    char* ws = (char*)d_ws;
    size_t off = 0;
    auto alloc = [&](size_t bytes) { void* p = ws + off; off += (bytes + 255) & ~(size_t)255; return p; };

    _Float16* h16  = (_Float16*)alloc((size_t)ROWS * Dv * 2);
    float*    hf   = (float*)   alloc((size_t)ROWS * Dv * 4);
    _Float16* WqT  = (_Float16*)alloc((size_t)Hv * HSv * Dv * 2);
    _Float16* WkT  = (_Float16*)alloc((size_t)Hv * HSv * Dv * 2);
    _Float16* WvT  = (_Float16*)alloc((size_t)Hv * HSv * Dv * 2);
    _Float16* WpT  = (_Float16*)alloc((size_t)Dv * Dv * 2);
    _Float16* W1T  = (_Float16*)alloc((size_t)Dv * Fv * 2);
    _Float16* W2T  = (_Float16*)alloc((size_t)Fv * Dv * 2);
    _Float16* q16  = (_Float16*)alloc((size_t)Bv * Hv * Tv * HSv * 2);
    _Float16* k16  = (_Float16*)alloc((size_t)Bv * Hv * Tv * HSv * 2);
    _Float16* vT16 = (_Float16*)alloc((size_t)Bv * Hv * HSv * Tv * 2);
    _Float16* o16  = (_Float16*)alloc((size_t)ROWS * Dv * 2);
    float*    x2f  = (float*)   alloc((size_t)ROWS * Dv * 4);
    float*    h2f  = (float*)   alloc((size_t)ROWS * Dv * 4);
    _Float16* h2h  = (_Float16*)alloc((size_t)ROWS * Dv * 2);
    _Float16* ff1  = (_Float16*)alloc((size_t)ROWS * Fv * 2);
    (void)ws_size; (void)n_in; (void)in_sizes; (void)out_size;

    // 1. LN1 -> h (f32 + f16)
    ln_kernel<<<ROWS, 256, 0, stream>>>(x, g1, be1, hf, h16);

    // 2. Weight convert + transpose to f16 N-major
    cvt_transpose<<<dim3(256, Hv), 256, 0, stream>>>(Wq, WqT, Dv, HSv);
    cvt_transpose<<<dim3(256, Hv), 256, 0, stream>>>(Wk, WkT, Dv, HSv);
    cvt_transpose<<<dim3(256, Hv), 256, 0, stream>>>(Wv, WvT, Dv, HSv);
    cvt_transpose<<<dim3(4096, 1), 256, 0, stream>>>(Wp, WpT, Dv, Dv);
    cvt_transpose<<<dim3(16384, 1), 256, 0, stream>>>(W1, W1T, Dv, Fv);
    cvt_transpose<<<dim3(16384, 1), 256, 0, stream>>>(W2, W2T, Fv, Dv);

    // 3. QKV projections (V transposed)
    qkv_kernel<<<dim3(Tv / 128, Bv * Hv), 256, 0, stream>>>(h16, WqT, WkT, WvT,
                                                            q16, k16, vT16);
    // 4. Causal flash attention -> o16 [B*T, D] (heads concatenated)
    attn_kernel<<<dim3(Tv / 128, Bv * Hv), 256, 0, stream>>>(q16, k16, vT16, o16);

    // 5. Output projection + residual (h): x2 = h + o @ Wp + bp
    gemm_kernel<false, true, false><<<dim3(Dv / 256, ROWS / 64), 256, 0, stream>>>(
        o16, WpT, bp, hf, x2f, nullptr, ROWS, Dv, Dv);

    // 6. LN2 -> h2 (f32 + f16)
    ln_kernel<<<ROWS, 256, 0, stream>>>(x2f, g2, be2, h2f, h2h);

    // 7. FFN1: relu(h2 @ W1 + b1) -> f16
    gemm_kernel<true, false, true><<<dim3(Fv / 256, ROWS / 64), 256, 0, stream>>>(
        h2h, W1T, b1, nullptr, nullptr, ff1, ROWS, Fv, Dv);

    // 8. FFN2 + residual (h2): out = h2 + ff1 @ W2 + b2
    gemm_kernel<false, true, false><<<dim3(Dv / 256, ROWS / 64), 256, 0, stream>>>(
        ff1, W2T, b2, h2f, out, nullptr, ROWS, Dv, Fv);
}
